// ExplainableVisionTransformer_74715251081368
// MI455X (gfx1250) — matrix operand order repaired and checked by reading, not compiled, not run on hardware
//
#include <hip/hip_runtime.h>
#include <hip/hip_bf16.h>
#include <math.h>

// ---------------------------------------------------------------------------
// CDNA5 (gfx1250) wave32 WMMA types
// ---------------------------------------------------------------------------
typedef __bf16 bf16_t;
typedef __attribute__((ext_vector_type(16))) __bf16 v16bf;
typedef __attribute__((ext_vector_type(8)))  __bf16 v8bf;
typedef __attribute__((ext_vector_type(2)))  __bf16 v2bf;
typedef __attribute__((ext_vector_type(8)))  float  v8f;

union Frag16 { v16bf v; v8bf h[2]; };
union CFrag8 { v8f   v; float e[8]; };

__device__ __forceinline__ unsigned short bfbits(float f) {
  union { float f; unsigned u; } c; c.f = f;
  unsigned r = c.u + 0x7FFFu + ((c.u >> 16) & 1u);   // round-to-nearest-even
  return (unsigned short)(r >> 16);
}

#if defined(__has_builtin)
#  if __has_builtin(__builtin_amdgcn_cvt_pk_bf16_f32)
#    define HAVE_CVT_PK_BF16 1
#  endif
#endif

// pack two fp32 -> packed bf16x2
__device__ __forceinline__ unsigned pack2bf(float lo, float hi) {
#ifdef HAVE_CVT_PK_BF16
  union { v2bf v; unsigned u; } t;
  t.v = __builtin_amdgcn_cvt_pk_bf16_f32(lo, hi);
  return t.u;
#else
  return (unsigned)bfbits(lo) | ((unsigned)bfbits(hi) << 16);
#endif
}

__device__ __forceinline__ float gelu_exact(float x) {
  return 0.5f * x * (1.0f + erff(x * 0.70710678118654752f));
}

// ---------------------------------------------------------------------------
// Generic tiled WMMA GEMM:  C[M,N] = epilogue(A[M,K] * B[K,N])
// FLAGS: 1=B stored [N,K] (transposed), 2=bias, 4=GELU, 8=residual, 16=scale
// Batched via grid.z: per-operand offset = (z/Hd)*s?0 + (z%Hd)*s?1
// Block = 256 threads = 8 wave32 waves; tile 128(M) x 64(N) x 32(K).
// Waves in 4(M)x2(N) grid; each wave owns a 32x32 output tile:
//   2 A-frags x 2 B-frags -> 4 v_wmma_f32_16x16x32_bf16 per K-step,
//   each fragment feeding two WMMAs (2x operand reuse vs 16x32 tiles).
// LDS: As[r][k], Bst[n][k] both K-contiguous, pitch 40 bf16 (80B) -> every
// fragment is two aligned ds_load_b128; staging stores are ds_store_b64.
// ---------------------------------------------------------------------------
template<int FLAGS>
__global__ __launch_bounds__(256)
void gemm_wmma_bf16(const float* __restrict__ A, const float* __restrict__ Bm,
                    const float* __restrict__ bias, const float* __restrict__ Res,
                    float* __restrict__ C,
                    int M, int N, int K, int lda, int ldb, int ldc,
                    int Hd,
                    long long sA0, long long sA1, long long sB0, long long sB1,
                    long long sC0, long long sC1, float scale)
{
  constexpr bool TRB   = (FLAGS & 1)  != 0;
  constexpr bool BIAS  = (FLAGS & 2)  != 0;
  constexpr bool GELU  = (FLAGS & 4)  != 0;
  constexpr bool RESID = (FLAGS & 8)  != 0;
  constexpr bool SCALE = (FLAGS & 16) != 0;

  constexpr int BM = 128, BN = 64, BK = 32, LDW = 40;  // 80B pitch, 16B aligned
  __shared__ __align__(16) bf16_t As [BM][LDW];
  __shared__ __align__(16) bf16_t Bst[BN][LDW];        // B transposed: [n][k]

  const int z  = blockIdx.z;
  const int zb = z / Hd, zh = z % Hd;
  const float* Ab = A  + zb * sA0 + zh * sA1;
  const float* Bb = Bm + zb * sB0 + zh * sB1;
  const long long cOff = zb * sC0 + zh * sC1;

  const int m0 = blockIdx.y * BM;
  const int n0 = blockIdx.x * BN;

  const int tid  = threadIdx.x;
  const int lane = tid & 31;
  const int wid  = tid >> 5;
  const int wm   = wid & 3;   // 32-row M sub-tile
  const int wn   = wid >> 2;  // 32-col N sub-tile

  const bool mFull = (m0 + BM) <= M;
  const bool nFull = (n0 + BN) <= N;
  const bool aVec = mFull && ((lda & 3) == 0) &&
                    ((((unsigned long long)(const void*)Ab) & 15ull) == 0);
  const bool bVec = TRB && nFull && ((ldb & 3) == 0) &&
                    ((((unsigned long long)(const void*)Bb) & 15ull) == 0);

  CFrag8 acc00, acc01, acc10, acc11;
#pragma unroll
  for (int i = 0; i < 8; ++i) {
    acc00.e[i] = 0.0f; acc01.e[i] = 0.0f; acc10.e[i] = 0.0f; acc11.e[i] = 0.0f;
  }

  const int kTiles = (K + BK - 1) / BK;
  for (int kt = 0; kt < kTiles; ++kt) {
    const int k0 = kt * BK;
    const bool kFull = (k0 + BK) <= K;

    // ---- stage A tile (128x32): float4 -> bf16x4 -> ds_store_b64 ----
    if (aVec && kFull) {
#pragma unroll
      for (int j = 0; j < 4; ++j) {
        int idx = tid + j * 256;          // 0..1023
        int r = idx >> 3;                 // 0..127
        int kq = (idx & 7) * 4;           // 0..28
        const float4 t = *(const float4*)(Ab + (long long)(m0 + r) * lda + (k0 + kq));
        *(uint2*)(&As[r][kq]) = make_uint2(pack2bf(t.x, t.y), pack2bf(t.z, t.w));
      }
    } else {
#pragma unroll
      for (int j = 0; j < 4; ++j) {
        int idx = tid + j * 256;
        int r = idx >> 3;
        int kq = (idx & 7) * 4;
        float v[4];
#pragma unroll
        for (int u = 0; u < 4; ++u) {
          int gm = m0 + r, gk = k0 + kq + u;
          v[u] = (gm < M && gk < K) ? Ab[(long long)gm * lda + gk] : 0.0f;
        }
        *(uint2*)(&As[r][kq]) = make_uint2(pack2bf(v[0], v[1]), pack2bf(v[2], v[3]));
      }
    }

    // ---- stage B tile (32x64) into Bst[n][k] ----
    if (TRB) {                            // B global layout [N,K]: K contiguous
      if (bVec && kFull) {
#pragma unroll
        for (int j = 0; j < 2; ++j) {
          int idx = tid + j * 256;
          int n = idx >> 3;
          int kq = (idx & 7) * 4;
          const float4 t = *(const float4*)(Bb + (long long)(n0 + n) * ldb + (k0 + kq));
          *(uint2*)(&Bst[n][kq]) = make_uint2(pack2bf(t.x, t.y), pack2bf(t.z, t.w));
        }
      } else {
#pragma unroll
        for (int j = 0; j < 2; ++j) {
          int idx = tid + j * 256;
          int n = idx >> 3;
          int kq = (idx & 7) * 4;
          float v[4];
#pragma unroll
          for (int u = 0; u < 4; ++u) {
            int gn = n0 + n, gk = k0 + kq + u;
            v[u] = (gn < N && gk < K) ? Bb[(long long)gn * ldb + gk] : 0.0f;
          }
          *(uint2*)(&Bst[n][kq]) = make_uint2(pack2bf(v[0], v[1]), pack2bf(v[2], v[3]));
        }
      }
    } else {                              // B global layout [K,N]: gather 4 k's
#pragma unroll
      for (int j = 0; j < 2; ++j) {
        int idx = tid + j * 256;
        int n = idx & 63;                 // lane-contiguous n -> coalesced loads
        int kq = (idx >> 6) * 4;
        float v[4];
#pragma unroll
        for (int u = 0; u < 4; ++u) {
          int gk = k0 + kq + u, gn = n0 + n;
          v[u] = (gk < K && gn < N) ? Bb[(long long)gk * ldb + gn] : 0.0f;
        }
        *(uint2*)(&Bst[n][kq]) = make_uint2(pack2bf(v[0], v[1]), pack2bf(v[2], v[3]));
      }
    }
    __syncthreads();

    // prefetch next A K-tile while this one computes
    if (kt + 1 < kTiles) {
      int pr = m0 + (tid >> 1);
      if (pr < M && (k0 + BK) < K)
        __builtin_prefetch(Ab + (long long)pr * lda + (k0 + BK), 0, 1);
    }

    // ---- fragments: two aligned ds_load_b128 each (ISA 7.12.2 layouts) ----
    Frag16 a0, a1;
    {
      int r   = wm * 32 + (lane & 15);
      int akb = (lane >> 4) * 8;          // K {0-7,16-23} / {8-15,24-31}
      a0.h[0] = *(const v8bf*)(&As[r][akb]);
      a0.h[1] = *(const v8bf*)(&As[r][16 + akb]);
      a1.h[0] = *(const v8bf*)(&As[r + 16][akb]);
      a1.h[1] = *(const v8bf*)(&As[r + 16][16 + akb]);
    }
    Frag16 b0, b1;
    {
      int nc  = wn * 32 + (lane & 15);
      int bkb = (lane >> 4) * 8;          // K {0-7,16-23} / {8-15,24-31}
      b0.h[0] = *(const v8bf*)(&Bst[nc][bkb]);
      b0.h[1] = *(const v8bf*)(&Bst[nc][16 + bkb]);
      b1.h[0] = *(const v8bf*)(&Bst[nc + 16][bkb]);
      b1.h[1] = *(const v8bf*)(&Bst[nc + 16][16 + bkb]);
    }

    acc00.v = __builtin_amdgcn_wmma_f32_16x16x32_bf16(false, a0.v, false, b0.v,
                                                      (short)0, acc00.v, false, false);
    acc01.v = __builtin_amdgcn_wmma_f32_16x16x32_bf16(false, a0.v, false, b1.v,
                                                      (short)0, acc01.v, false, false);
    acc10.v = __builtin_amdgcn_wmma_f32_16x16x32_bf16(false, a1.v, false, b0.v,
                                                      (short)0, acc10.v, false, false);
    acc11.v = __builtin_amdgcn_wmma_f32_16x16x32_bf16(false, a1.v, false, b1.v,
                                                      (short)0, acc11.v, false, false);
    __syncthreads();
  }

  // ---- epilogue + store (C/D: M = vgpr + 8*(lane>=16), N = lane&15) ----
  const int rb0 = m0 + wm * 32 + (lane >> 4) * 8;
  const int cn0 = n0 + wn * 32 + (lane & 15);
#pragma unroll
  for (int mt = 0; mt < 2; ++mt) {
#pragma unroll
    for (int t = 0; t < 2; ++t) {
      CFrag8& acc = mt ? (t ? acc11 : acc10) : (t ? acc01 : acc00);
      int cn = cn0 + t * 16;
#pragma unroll
      for (int vv = 0; vv < 8; ++vv) {
        int r = rb0 + mt * 16 + vv;
        if (r < M && cn < N) {
          float x = acc.e[vv];
          if (SCALE) x *= scale;
          if (BIAS)  x += bias[cn];
          if (GELU)  x = gelu_exact(x);
          long long ci = cOff + (long long)r * ldc + cn;
          if (RESID) x += Res[ci];
          C[ci] = x;
        }
      }
    }
  }
}

// ---------------------------------------------------------------------------
// im2col for 16x16 stride-16 conv: out[b*196+py*14+px][ch*256+ph*16+pw]
// ---------------------------------------------------------------------------
__global__ void im2col_kernel(const float* __restrict__ x, float* __restrict__ out)
{
  long long i = (long long)blockIdx.x * blockDim.x + threadIdx.x;
  const long long total = 32LL * 196 * 768;
  if (i >= total) return;
  int c = (int)(i % 768);
  long long rr = i / 768;
  int b = (int)(rr / 196);
  int p = (int)(rr % 196);
  int py = p / 14, px = p % 14;
  int ch = c >> 8, ph = (c >> 4) & 15, pw = c & 15;
  int iy = py * 16 + ph, ix = px * 16 + pw;
  out[i] = x[((long long)b * 3 + ch) * 224 * 224 + (long long)iy * 224 + ix];
}

// tok[b,n,d] = (n==0 ? cls[d] : patches[b,n-1,d]) + pos[n,d]
__global__ void build_tok_kernel(const float* __restrict__ pe_out,
                                 const float* __restrict__ cls_token,
                                 const float* __restrict__ pos,
                                 float* __restrict__ tok)
{
  long long i = (long long)blockIdx.x * blockDim.x + threadIdx.x;
  const long long total = 32LL * 197 * 768;
  if (i >= total) return;
  int d = (int)(i % 768);
  long long rn = i / 768;
  int b = (int)(rn / 197);
  int n = (int)(rn % 197);
  float v = (n == 0) ? cls_token[d]
                     : pe_out[((long long)b * 196 + (n - 1)) * 768 + d];
  tok[i] = v + pos[(long long)n * 768 + d];
}

// ---------------------------------------------------------------------------
// LayerNorm over 768-wide rows; one 256-thread block (8 waves) per row
// ---------------------------------------------------------------------------
__global__ __launch_bounds__(256)
void layernorm768(const float* __restrict__ X, const float* __restrict__ w,
                  const float* __restrict__ bb, float* __restrict__ Y)
{
  int row = blockIdx.x;
  const float* x = X + (long long)row * 768;
  float* y = Y + (long long)row * 768;
  int tid = threadIdx.x, wid = tid >> 5, lane = tid & 31;

  float vals[3]; float s = 0.f, s2 = 0.f;
#pragma unroll
  for (int j = 0; j < 3; ++j) {
    float v = x[tid + j * 256]; vals[j] = v; s += v; s2 += v * v;
  }
#pragma unroll
  for (int o = 16; o > 0; o >>= 1) { s += __shfl_xor(s, o); s2 += __shfl_xor(s2, o); }
  __shared__ float red0[8], red1[8];
  if (lane == 0) { red0[wid] = s; red1[wid] = s2; }
  __syncthreads();
  float ts = 0.f, ts2 = 0.f;
#pragma unroll
  for (int i = 0; i < 8; ++i) { ts += red0[i]; ts2 += red1[i]; }
  float mu  = ts * (1.0f / 768.0f);
  float var = ts2 * (1.0f / 768.0f) - mu * mu;
  float inv = rsqrtf(var + 1e-5f);
#pragma unroll
  for (int j = 0; j < 3; ++j) {
    int c = tid + j * 256;
    y[c] = (vals[j] - mu) * inv * w[c] + bb[c];
  }
}

// Fused final LN(fnorm) -> LN(hln) on the cls token only (one block per batch)
__global__ __launch_bounds__(256)
void cls_double_ln(const float* __restrict__ tok,
                   const float* __restrict__ fw, const float* __restrict__ fb,
                   const float* __restrict__ hw, const float* __restrict__ hb,
                   float* __restrict__ out)
{
  int b = blockIdx.x;
  const float* x = tok + (long long)b * 197 * 768;  // row n==0
  __shared__ float tmp[768];
  __shared__ float red0[8], red1[8];
  int tid = threadIdx.x, wid = tid >> 5, lane = tid & 31;

  float vals[3]; float s = 0.f, s2 = 0.f;
#pragma unroll
  for (int j = 0; j < 3; ++j) { float v = x[tid + j * 256]; vals[j] = v; s += v; s2 += v * v; }
#pragma unroll
  for (int o = 16; o > 0; o >>= 1) { s += __shfl_xor(s, o); s2 += __shfl_xor(s2, o); }
  if (lane == 0) { red0[wid] = s; red1[wid] = s2; }
  __syncthreads();
  float ts = 0.f, ts2 = 0.f;
#pragma unroll
  for (int i = 0; i < 8; ++i) { ts += red0[i]; ts2 += red1[i]; }
  float mu = ts * (1.0f / 768.0f), var = ts2 * (1.0f / 768.0f) - mu * mu;
  float inv = rsqrtf(var + 1e-5f);
#pragma unroll
  for (int j = 0; j < 3; ++j) {
    int c = tid + j * 256;
    tmp[c] = (vals[j] - mu) * inv * fw[c] + fb[c];
  }
  __syncthreads();

  s = 0.f; s2 = 0.f;
#pragma unroll
  for (int j = 0; j < 3; ++j) { float v = tmp[tid + j * 256]; vals[j] = v; s += v; s2 += v * v; }
#pragma unroll
  for (int o = 16; o > 0; o >>= 1) { s += __shfl_xor(s, o); s2 += __shfl_xor(s2, o); }
  __syncthreads();                 // pass-1 red reads done; safe to overwrite
  if (lane == 0) { red0[wid] = s; red1[wid] = s2; }
  __syncthreads();
  ts = 0.f; ts2 = 0.f;
#pragma unroll
  for (int i = 0; i < 8; ++i) { ts += red0[i]; ts2 += red1[i]; }
  mu = ts * (1.0f / 768.0f); var = ts2 * (1.0f / 768.0f) - mu * mu;
  inv = rsqrtf(var + 1e-5f);
#pragma unroll
  for (int j = 0; j < 3; ++j) {
    int c = tid + j * 256;
    out[(long long)b * 768 + c] = (vals[j] - mu) * inv * hw[c] + hb[c];
  }
}

// ---------------------------------------------------------------------------
// Softmax over rows of length 197 (one wave32 per row, 4 rows per block)
// ---------------------------------------------------------------------------
__global__ __launch_bounds__(128)
void softmax_rows197(float* __restrict__ S, int nrows)
{
  int row = blockIdx.x * 4 + (threadIdx.x >> 5);
  if (row >= nrows) return;
  int lane = threadIdx.x & 31;
  float* p = S + (long long)row * 197;
  float v[7]; float mx = -3.0e38f;
#pragma unroll
  for (int j = 0; j < 7; ++j) {
    int c = lane + j * 32;
    v[j] = (c < 197) ? p[c] : -3.0e38f;
    mx = fmaxf(mx, v[j]);
  }
#pragma unroll
  for (int o = 16; o > 0; o >>= 1) mx = fmaxf(mx, __shfl_xor(mx, o));
  float sum = 0.f;
#pragma unroll
  for (int j = 0; j < 7; ++j) { v[j] = __expf(v[j] - mx); sum += v[j]; }
#pragma unroll
  for (int o = 16; o > 0; o >>= 1) sum += __shfl_xor(sum, o);
  float inv = 1.0f / sum;
#pragma unroll
  for (int j = 0; j < 7; ++j) { int c = lane + j * 32; if (c < 197) p[c] = v[j] * inv; }
}

// ---------------------------------------------------------------------------
// Host-side orchestration
// ---------------------------------------------------------------------------
extern "C" void kernel_launch(void* const* d_in, const int* in_sizes, int n_in,
                              void* d_out, int out_size, void* d_ws, size_t ws_size,
                              hipStream_t stream)
{
  (void)in_sizes; (void)n_in; (void)out_size; (void)ws_size;

  const float* x       = (const float*)d_in[0];
  const float* conv_w  = (const float*)d_in[1];
  const float* conv_b  = (const float*)d_in[2];
  const float* cls_tok = (const float*)d_in[3];
  const float* pos     = (const float*)d_in[4];
  const float* ln1_w   = (const float*)d_in[5];
  const float* ln1_b   = (const float*)d_in[6];
  const float* wq      = (const float*)d_in[7];
  const float* wk      = (const float*)d_in[8];
  const float* wv      = (const float*)d_in[9];
  const float* wo_w    = (const float*)d_in[10];
  const float* wo_b    = (const float*)d_in[11];
  const float* ln2_w   = (const float*)d_in[12];
  const float* ln2_b   = (const float*)d_in[13];
  const float* mlp1_w  = (const float*)d_in[14];
  const float* mlp1_b  = (const float*)d_in[15];
  const float* mlp2_w  = (const float*)d_in[16];
  const float* mlp2_b  = (const float*)d_in[17];
  const float* fnorm_w = (const float*)d_in[18];
  const float* fnorm_b = (const float*)d_in[19];
  const float* hln_w   = (const float*)d_in[20];
  const float* hln_b   = (const float*)d_in[21];
  const float* h1_w    = (const float*)d_in[22];
  const float* h1_b    = (const float*)d_in[23];
  const float* h2_w    = (const float*)d_in[24];
  const float* h2_b    = (const float*)d_in[25];

  const int Bn = 32, Nn = 197, Dd = 768, Hh = 12, DKd = 64, MLPd = 3072,
            DEPTH = 12, NP = 196;
  const long long T   = (long long)Bn * Nn * Dd;          // 4,841,472 floats
  const long long ATT = (long long)Bn * Hh * Nn * Nn;     // 14,902,656
  const long long M1  = (long long)Bn * Nn * MLPd;        // 19,365,888

  float* ws   = (float*)d_ws;
  float* tok  = ws;
  float* hbuf = ws + T;
  float* qb   = ws + 2 * T;
  float* kb   = ws + 3 * T;
  float* vb   = ws + 4 * T;
  float* ctxb = ws + 5 * T;
  float* attn = ws + 6 * T;
  float* m1   = attn + ATT;
  float* clsb = m1 + M1;
  float* h1o  = clsb + 32LL * 768;
  float* im2c = m1;     // reused: only live before the layer loop
  float* peo  = attn;   // reused: only live before the layer loop

  dim3 blk(256);
  const long long Z6 = 0;

  // 1) patch embed = im2col + GEMM(+bias,+GELU); conv_w is [N=768,K=768] -> transB
  {
    long long total = (long long)Bn * NP * Dd;
    im2col_kernel<<<dim3((unsigned)((total + 255) / 256)), blk, 0, stream>>>(x, im2c);
    int M = Bn * NP, N = Dd, K = Dd;
    dim3 grid((N + 63) / 64, (M + 127) / 128, 1);
    gemm_wmma_bf16<1 | 2 | 4><<<grid, blk, 0, stream>>>(
        im2c, conv_w, conv_b, nullptr, peo, M, N, K, K, K, N,
        1, Z6, Z6, Z6, Z6, Z6, Z6, 1.0f);
  }
  // 2) assemble tokens (+cls, +pos)
  build_tok_kernel<<<dim3((unsigned)((T + 255) / 256)), blk, 0, stream>>>(peo, cls_tok, pos, tok);

  const int Mtok = Bn * Nn;                                  // 6304
  dim3 gridD((Dd + 63) / 64, (Mtok + 127) / 128, 1);         // 12 x 50
  dim3 gridM((MLPd + 63) / 64, (Mtok + 127) / 128, 1);       // 48 x 50
  const long long sQ0 = (long long)Nn * Dd;                  // per-batch stride (q/k/v/ctx)
  const long long sQ1 = 64;                                  // per-head stride
  const long long sS0 = (long long)Hh * Nn * Nn;             // per-batch stride (scores)
  const long long sS1 = (long long)Nn * Nn;                  // per-head stride

  // 3) transformer blocks
  for (int i = 0; i < DEPTH; ++i) {
    const float* Wq = wq + (long long)i * Dd * Dd;
    const float* Wk = wk + (long long)i * Dd * Dd;
    const float* Wv = wv + (long long)i * Dd * Dd;

    layernorm768<<<dim3(Mtok), blk, 0, stream>>>(tok, ln1_w + i * Dd, ln1_b + i * Dd, hbuf);

    gemm_wmma_bf16<0><<<gridD, blk, 0, stream>>>(hbuf, Wq, nullptr, nullptr, qb,
        Mtok, Dd, Dd, Dd, Dd, Dd, 1, Z6, Z6, Z6, Z6, Z6, Z6, 1.0f);
    gemm_wmma_bf16<0><<<gridD, blk, 0, stream>>>(hbuf, Wk, nullptr, nullptr, kb,
        Mtok, Dd, Dd, Dd, Dd, Dd, 1, Z6, Z6, Z6, Z6, Z6, Z6, 1.0f);
    gemm_wmma_bf16<0><<<gridD, blk, 0, stream>>>(hbuf, Wv, nullptr, nullptr, vb,
        Mtok, Dd, Dd, Dd, Dd, Dd, 1, Z6, Z6, Z6, Z6, Z6, Z6, 1.0f);

    // scores[b,h] = q[b,:,h,:] @ k[b,:,h,:]^T * 1/8   (batched over z = b*12+h)
    {
      dim3 grid((Nn + 63) / 64, (Nn + 127) / 128, Bn * Hh);
      gemm_wmma_bf16<1 | 16><<<grid, blk, 0, stream>>>(qb, kb, nullptr, nullptr, attn,
          Nn, Nn, DKd, Dd, Dd, Nn, Hh, sQ0, sQ1, sQ0, sQ1, sS0, sS1, 0.125f);
    }
    {
      int nrows = Bn * Hh * Nn;
      softmax_rows197<<<dim3((nrows + 3) / 4), dim3(128), 0, stream>>>(attn, nrows);
    }
    // ctx[b,:,h,:] = attn[b,h] @ v[b,:,h,:]   (K=197 zero-padded in the K-loop)
    {
      dim3 grid((DKd + 63) / 64, (Nn + 127) / 128, Bn * Hh);
      gemm_wmma_bf16<0><<<grid, blk, 0, stream>>>(attn, vb, nullptr, nullptr, ctxb,
          Nn, DKd, Nn, Nn, Dd, Dd, Hh, sS0, sS1, sQ0, sQ1, sQ0, sQ1, 1.0f);
    }
    // out proj + bias + residual (in place into tok)
    gemm_wmma_bf16<2 | 8><<<gridD, blk, 0, stream>>>(ctxb, wo_w + (long long)i * Dd * Dd,
        wo_b + i * Dd, tok, tok,
        Mtok, Dd, Dd, Dd, Dd, Dd, 1, Z6, Z6, Z6, Z6, Z6, Z6, 1.0f);

    layernorm768<<<dim3(Mtok), blk, 0, stream>>>(tok, ln2_w + i * Dd, ln2_b + i * Dd, hbuf);

    // MLP1 + bias + GELU
    gemm_wmma_bf16<2 | 4><<<gridM, blk, 0, stream>>>(hbuf, mlp1_w + (long long)i * Dd * MLPd,
        mlp1_b + (long long)i * MLPd, nullptr, m1,
        Mtok, MLPd, Dd, Dd, MLPd, MLPd, 1, Z6, Z6, Z6, Z6, Z6, Z6, 1.0f);
    // MLP2 + bias + residual (in place into tok)
    gemm_wmma_bf16<2 | 8><<<gridD, blk, 0, stream>>>(m1, mlp2_w + (long long)i * MLPd * Dd,
        mlp2_b + i * Dd, tok, tok,
        Mtok, Dd, MLPd, MLPd, Dd, Dd, 1, Z6, Z6, Z6, Z6, Z6, Z6, 1.0f);
  }

  // 4) head: LN(fnorm) -> LN(hln) on cls, then MLP head
  cls_double_ln<<<dim3(Bn), blk, 0, stream>>>(tok, fnorm_w, fnorm_b, hln_w, hln_b, clsb);
  {
    dim3 grid((MLPd + 63) / 64, 1, 1);
    gemm_wmma_bf16<2 | 4><<<grid, blk, 0, stream>>>(clsb, h1_w, h1_b, nullptr, h1o,
        Bn, MLPd, Dd, Dd, MLPd, MLPd, 1, Z6, Z6, Z6, Z6, Z6, Z6, 1.0f);
  }
  {
    dim3 grid(1, 1, 1);
    gemm_wmma_bf16<2><<<grid, blk, 0, stream>>>(h1o, h2_w, h2_b, nullptr, (float*)d_out,
        Bn, 2, MLPd, MLPd, 2, 2, 1, Z6, Z6, Z6, Z6, Z6, Z6, 1.0f);
  }
}
